// Attention_84146999263843
// MI455X (gfx1250) — matrix operand order repaired and checked by reading, not compiled, not run on hardware
//
#include <hip/hip_runtime.h>
#include <hip/hip_bf16.h>

// ---------------------------------------------------------------------------
// Masked attention forward for MI455X (gfx1250), wave32 + WMMA f16->f32.
// B=4, L=2048, D_IN=D_MODEL=1024.  ~120 GFLOP vs ~80MB HBM -> compute bound:
// all five matmuls (QKV proj, QK^T, PV) run on v_wmma_f32_16x16x32_f16.
// Score row-block S[16][2048] lives in LDS (64KB of the 320KB WGP LDS) ->
// exact single-pass softmax matching masked_fill(-10000) semantics.
// This revision targets latency hiding via OCCUPANCY + ILP instead of a
// register-resident A tile: capped unrolling stops the compiler hoisting all
// 32 A fragments (~256 VGPRs, s_set_vgpr_msb, ~2 waves/SIMD), and each wave
// now computes TWO independent output tiles sharing one A fragment.
// ---------------------------------------------------------------------------

typedef _Float16 half_t;
typedef __attribute__((ext_vector_type(16))) _Float16 v16h;
typedef __attribute__((ext_vector_type(8)))  _Float16 v8h;
typedef __attribute__((ext_vector_type(4)))  _Float16 v4h;
typedef __attribute__((ext_vector_type(8)))  float    v8f;
typedef __attribute__((ext_vector_type(4)))  float    v4f;

#define BATCH   4
#define LSEQ    2048
#define DMODEL  1024
#define NEGINF  (-10000.0f)
#define QSCALE  0.03125f   // 1/sqrt(1024)

union V16U { v16h v; v8h h[2]; };

// A-matrix fragment (16x32 f16, M x K) from row-major [.., ld] storage.
// lanes 0-15 -> M=lane, K = k0..k0+15;
// lanes 16-31 -> M=lane-16, K = k0+16..k0+31 (matches load_fragBT).
__device__ __forceinline__ v16h load_fragA(const half_t* base, int ld, int k0, int lane) {
    int row = lane & 15;
    int hi  = (lane >> 4) & 1;
    const half_t* p = base + (size_t)row * ld + k0 + hi * 16;
    V16U u;
    u.v = *(const v16h*)(p);          // K = k0 + 16*hi + [0..15]
    return u.v;
}

// B-matrix fragment (32x16 f16, K x N) where the source is stored
// "transposed" row-major as BT[n][k] (so per-lane K is contiguous 32B).
// Order used here (same as load_fragA): lane -> column N = lane&15; element e -> K = k0 + 16*hi + e.
__device__ __forceinline__ v16h load_fragBT(const half_t* bt, int ld, int n0, int k0, int lane) {
    int n  = lane & 15;
    int hi = (lane >> 4) & 1;
    return *(const v16h*)(bt + (size_t)(n0 + n) * ld + k0 + hi * 16);
}

__device__ __forceinline__ v8f wmma_f16(v16h a, v16h b, v8f c) {
    return __builtin_amdgcn_wmma_f32_16x16x32_f16(
        /*neg_a=*/false, a, /*neg_b=*/false, b,
        /*c_mod=*/(short)0, c, /*reuse_a=*/false, /*reuse_b=*/false);
}

// ---------------------------------------------------------------------------
// Kernel 1: weight convert f32 -> f16 with transpose: wt[n][k] = w[k][n].
// ---------------------------------------------------------------------------
__global__ __launch_bounds__(256) void wcvt_transpose(const float* __restrict__ w,
                                                      half_t* __restrict__ wt) {
    int i = blockIdx.x * 256 + threadIdx.x;   // 0 .. 1M-1
    int k = i >> 10;
    int n = i & 1023;
    wt[(size_t)n * DMODEL + k] = (half_t)w[i];
}

// ---------------------------------------------------------------------------
// Kernel 2: QKV projection.  One WG (8 waves) per 16-row tile of B*L rows.
// Waves process PAIRS of the 192 output tiles (t, t+96) sharing one A frag:
// two independent WMMA chains + 4 loads in flight per A read.
// ---------------------------------------------------------------------------
__device__ __forceinline__ void store_proj(int mat, int n0, v8f acc,
                                           int b, int key0, int r0, int lane15, int hi,
                                           half_t* __restrict__ Q, half_t* __restrict__ K,
                                           half_t* __restrict__ VT) {
    if (mat == 2) {
        // VT[b][n][key]: per lane, col fixed, 8 contiguous keys -> one 16B store
        v8h pk;
        for (int j = 0; j < 8; ++j) pk[j] = (half_t)acc[j];
        *(v8h*)(VT + ((size_t)b * DMODEL + n0 + lane15) * LSEQ + key0 + hi * 8) = pk;
    } else {
        half_t* dst = (mat == 0) ? Q : K;
        const float sc = (mat == 0) ? QSCALE : 1.0f;
        for (int j = 0; j < 8; ++j)
            dst[(size_t)(r0 + hi * 8 + j) * DMODEL + n0 + lane15] = (half_t)(acc[j] * sc);
    }
}

__global__ __launch_bounds__(256) void proj_qkv(const float* __restrict__ x,
                                                const half_t* __restrict__ wqt,
                                                const half_t* __restrict__ wkt,
                                                const half_t* __restrict__ wvt,
                                                half_t* __restrict__ Q,
                                                half_t* __restrict__ K,
                                                half_t* __restrict__ VT) {
    __shared__ __align__(32) half_t xs[16 * DMODEL];   // 32 KB

    const int mt = blockIdx.x;          // 0..511
    const int r0 = mt << 4;             // tile never crosses batch (2048 % 16 == 0)
    const float* src = x + (size_t)r0 * DMODEL;

    for (int i = threadIdx.x; i < (16 * DMODEL) / 4; i += 256) {
        v4f f = ((const v4f*)src)[i];
        v4h h;
        h[0] = (half_t)f[0]; h[1] = (half_t)f[1];
        h[2] = (half_t)f[2]; h[3] = (half_t)f[3];
        ((v4h*)xs)[i] = h;
    }
    __syncthreads();

    const int wave   = threadIdx.x >> 5;
    const int lane   = threadIdx.x & 31;
    const int lane15 = lane & 15;
    const int hi     = (lane >> 4) & 1;
    const int b      = r0 >> 11;
    const int key0   = r0 & (LSEQ - 1);

    for (int t = wave; t < 96; t += 8) {
        const int t2   = t + 96;
        const int mat1 = t >> 6,  n01 = (t  & 63) << 4;
        const int mat2 = t2 >> 6, n02 = (t2 & 63) << 4;
        const half_t* w1 = (mat1 == 0) ? wqt : (mat1 == 1) ? wkt : wvt;
        const half_t* w2 = (mat2 == 0) ? wqt : (mat2 == 1) ? wkt : wvt;

        v8f acc1, acc2;
        for (int j = 0; j < 8; ++j) { acc1[j] = 0.0f; acc2[j] = 0.0f; }

        #pragma unroll 2
        for (int k0 = 0; k0 < DMODEL; k0 += 32) {
            v16h a  = load_fragA(xs, DMODEL, k0, lane);
            v16h b1 = load_fragBT(w1, DMODEL, n01, k0, lane);
            v16h b2 = load_fragBT(w2, DMODEL, n02, k0, lane);
            acc1 = wmma_f16(a, b1, acc1);
            acc2 = wmma_f16(a, b2, acc2);
        }
        store_proj(mat1, n01, acc1, b, key0, r0, lane15, hi, Q, K, VT);
        store_proj(mat2, n02, acc2, b, key0, r0, lane15, hi, Q, K, VT);
    }
}

// ---------------------------------------------------------------------------
// Kernel 3: attention.  One WG (8 waves) per (batch, 16-query tile).
// Phase 1: each wave computes a PAIR of adjacent 16-key score tiles sharing
//          one Q fragment (guard is wave-uniform -> EXEC all-ones for WMMA).
// Phase 2: exact softmax (shfl_xor, wave32).
// Phase 3: O = P V, waves split the 1024 output columns.
// ---------------------------------------------------------------------------
__global__ __launch_bounds__(256) void attn_fused(const half_t* __restrict__ Q,
                                                  const half_t* __restrict__ K,
                                                  const half_t* __restrict__ VT,
                                                  const int* __restrict__ pad,
                                                  float* __restrict__ out) {
    __shared__ __align__(32) half_t qs[16 * DMODEL];  // 32 KB
    __shared__ __align__(32) half_t S[16 * LSEQ];     // 64 KB score/prob tile
    __shared__ float rowsum[16];

    const int b  = blockIdx.x >> 7;
    const int qt = blockIdx.x & 127;
    const int q0 = qt << 4;

    const half_t* qsrc = Q + ((size_t)b * LSEQ + q0) * DMODEL;
    for (int i = threadIdx.x; i < (16 * DMODEL) / 8; i += 256)
        ((v8h*)qs)[i] = ((const v8h*)qsrc)[i];

    v8h negv; for (int j = 0; j < 8; ++j) negv[j] = (half_t)NEGINF;
    for (int i = threadIdx.x; i < (16 * LSEQ) / 8; i += 256)
        ((v8h*)S)[i] = negv;
    __syncthreads();

    const int wave   = threadIdx.x >> 5;
    const int lane   = threadIdx.x & 31;
    const int lane15 = lane & 15;
    const int hi     = (lane >> 4) & 1;

    // ---- Phase 1: score tiles intersecting the causal region ----
    const half_t* kb   = K + (size_t)b * LSEQ * DMODEL;  // [key][d] == B^T layout
    const int*    padb = pad + b * LSEQ;

    for (int p = wave; 2 * p <= qt; p += 8) {
        const int  nk1 = 2 * p;
        const int  k0a = nk1 << 4;
        const int  k0b = k0a + 16;
        const bool do2 = (nk1 + 1) <= qt;          // wave-uniform

        v8f acc1, acc2;
        for (int j = 0; j < 8; ++j) { acc1[j] = 0.0f; acc2[j] = 0.0f; }

        #pragma unroll 2
        for (int d0 = 0; d0 < DMODEL; d0 += 32) {
            v16h a  = load_fragA(qs, DMODEL, d0, lane);
            v16h b1 = load_fragBT(kb, DMODEL, k0a, d0, lane);
            acc1 = wmma_f16(a, b1, acc1);
            if (do2) {
                v16h b2 = load_fragBT(kb, DMODEL, k0b, d0, lane);
                acc2 = wmma_f16(a, b2, acc2);
            }
        }

        const int qrow = q0 + hi * 8;
        {
            const int  kcol = k0a + lane15;
            const bool pd   = (padb[kcol] != 0);
            for (int j = 0; j < 8; ++j) {
                float s = acc1[j];
                if (pd || (kcol > qrow + j)) s = NEGINF;
                S[(hi * 8 + j) * LSEQ + kcol] = (half_t)s;
            }
        }
        if (do2) {
            const int  kcol = k0b + lane15;
            const bool pd   = (padb[kcol] != 0);
            for (int j = 0; j < 8; ++j) {
                float s = acc2[j];
                if (pd || (kcol > qrow + j)) s = NEGINF;
                S[(hi * 8 + j) * LSEQ + kcol] = (half_t)s;
            }
        }
    }
    __syncthreads();

    // ---- Phase 2: exact softmax over full key axis ----
    {
        const int row = threadIdx.x >> 4;
        const int l16 = threadIdx.x & 15;
        half_t* srow = S + (size_t)row * LSEQ;

        float m = -3.0e38f;
        for (int c = l16; c < LSEQ; c += 16)
            m = fmaxf(m, (float)srow[c]);
        for (int mk = 1; mk < 16; mk <<= 1)
            m = fmaxf(m, __shfl_xor(m, mk, 16));

        float sum = 0.0f;
        for (int c = l16; c < LSEQ; c += 16) {
            float e = __expf((float)srow[c] - m);
            srow[c] = (half_t)e;            // S now holds unnormalized P (f16)
            sum += e;
        }
        for (int mk = 1; mk < 16; mk <<= 1)
            sum += __shfl_xor(sum, mk, 16);
        if (l16 == 0) rowsum[row] = sum;
    }
    __syncthreads();

    // ---- Phase 3: O = P * V ; wave w owns output columns [128w, 128w+128) ----
    const half_t* vtb = VT + (size_t)b * DMODEL * LSEQ;  // [n][key]
    const int nbase = wave * 128;

    v8f acc[8];
    for (int t = 0; t < 8; ++t)
        for (int j = 0; j < 8; ++j) acc[t][j] = 0.0f;

    #pragma unroll 1
    for (int k0 = 0; k0 < LSEQ; k0 += 32) {
        // stream-ahead hint on the V rows this wave owns (speculative, OOB-safe)
        __builtin_prefetch(vtb + (size_t)(nbase + lane * 4) * LSEQ + k0 + 256, 0, 0);

        v16h a = load_fragA(S, LSEQ, k0, lane);          // P from LDS
        #pragma unroll
        for (int t = 0; t < 8; ++t) {
            v16h bm = load_fragBT(vtb, LSEQ, nbase + t * 16, k0, lane);
            acc[t] = wmma_f16(a, bm, acc[t]);
        }
    }

    for (int t = 0; t < 8; ++t) {
        const int n = nbase + t * 16 + lane15;
        for (int j = 0; j < 8; ++j) {
            const int r = hi * 8 + j;
            out[((size_t)b * LSEQ + q0 + r) * DMODEL + n] = acc[t][j] / rowsum[r];
        }
    }
}

// ---------------------------------------------------------------------------
// Launcher.  Workspace layout (f16 elements), total ~54 MB.
// ---------------------------------------------------------------------------
extern "C" void kernel_launch(void* const* d_in, const int* in_sizes, int n_in,
                              void* d_out, int out_size, void* d_ws, size_t ws_size,
                              hipStream_t stream) {
    const float* x   = (const float*)d_in[0];
    const int*   pm  = (const int*)d_in[1];
    const float* wq  = (const float*)d_in[2];
    const float* wk  = (const float*)d_in[3];
    const float* wv  = (const float*)d_in[4];
    float*       out = (float*)d_out;

    half_t* wqt = (half_t*)d_ws;
    half_t* wkt = wqt + (size_t)DMODEL * DMODEL;
    half_t* wvt = wkt + (size_t)DMODEL * DMODEL;
    half_t* Q   = wvt + (size_t)DMODEL * DMODEL;
    half_t* Kp  = Q   + (size_t)BATCH * LSEQ * DMODEL;
    half_t* VT  = Kp  + (size_t)BATCH * LSEQ * DMODEL;

    wcvt_transpose<<<(DMODEL * DMODEL) / 256, 256, 0, stream>>>(wq, wqt);
    wcvt_transpose<<<(DMODEL * DMODEL) / 256, 256, 0, stream>>>(wk, wkt);
    wcvt_transpose<<<(DMODEL * DMODEL) / 256, 256, 0, stream>>>(wv, wvt);

    proj_qkv<<<(BATCH * LSEQ) / 16, 256, 0, stream>>>(x, wqt, wkt, wvt, Q, Kp, VT);

    attn_fused<<<BATCH * (LSEQ / 16), 256, 0, stream>>>(Q, Kp, VT, pm, out);
}
